// FoldUnfoldModule_82592221102582
// MI455X (gfx1250) — compile-verified, matrix operand order checked by reading, not run on hardware
//
#include <hip/hip_runtime.h>
#include <math.h>

// ---------------------------------------------------------------------------
// Problem constants
// ---------------------------------------------------------------------------
#define BB   64
#define CC   512
#define HH   26
#define WW   26
#define LL   (HH * WW)          // 676
#define CQ   128                // C/4
#define BN_EPS 1e-5f
#define LN_EPS 1e-6f
#define NEG_SLOPE 0.01f

// WMMA types (gfx1250 / CDNA5, wave32)
typedef __attribute__((ext_vector_type(16))) __bf16 v16bf;
typedef __attribute__((ext_vector_type(8)))  __bf16 v8bf;
typedef __attribute__((ext_vector_type(8)))  float  v8f;
// TDM descriptor vector types (probe-verified builtin signatures)
typedef unsigned int v4u  __attribute__((ext_vector_type(4)));
typedef int          v4i_ __attribute__((ext_vector_type(4)));
typedef int          v8i_ __attribute__((ext_vector_type(8)));

// ---------------------------------------------------------------------------
// Kernel 1: self-similarity  sim[b, o, p] = sum_c x[b,c,p] * x[b,c,neigh(p,o)]
// ---------------------------------------------------------------------------
__global__ __launch_bounds__(256)
void selfsim_kernel(const float* __restrict__ x, float* __restrict__ sim)
{
    int t = blockIdx.x * blockDim.x + threadIdx.x;   // 0 .. B*L-1
    if (t >= BB * LL) return;
    int b = t / LL, p = t % LL;
    int h = p / WW, w = p % WW;

    const float* xb = x + (long)b * CC * LL;

    float acc[8];
    int   nb[8];
    int   ok[8];
#pragma unroll
    for (int o = 1; o < 9; ++o) {
        int di = o / 3 - 1, dj = o % 3 - 1;
        int hh = h + di, ww = w + dj;
        ok[o - 1] = (hh >= 0 && hh < HH && ww >= 0 && ww < WW);
        nb[o - 1] = hh * WW + ww;
        acc[o - 1] = 0.f;
    }

    for (int c = 0; c < CC; ++c) {
        const float* row = xb + (long)c * LL;
        float xc = row[p];
#pragma unroll
        for (int o = 0; o < 8; ++o)
            if (ok[o]) acc[o] += xc * row[nb[o]];
    }

    float* sb = sim + (long)b * 9 * LL;
    sb[p] = 0.f;
#pragma unroll
    for (int o = 0; o < 8; ++o) sb[(o + 1) * LL + p] = acc[o];
}

// ---------------------------------------------------------------------------
// Kernel 2: 3x3 conv (9 -> 512) + BN1 + LeakyReLU, then xm = m * x
// sim[b] (24 KB contiguous) is staged into LDS by the Tensor Data Mover.
// ---------------------------------------------------------------------------
__global__ __launch_bounds__(256)
void conv_bn_mul_kernel(const float* __restrict__ sim,
                        const float* __restrict__ conv_w,
                        const float* __restrict__ g,  const float* __restrict__ bta,
                        const float* __restrict__ mu, const float* __restrict__ var,
                        const float* __restrict__ x,  float* __restrict__ xm)
{
    int bo = blockIdx.x;
    int b  = bo >> 9;           // /512
    int o  = bo & 511;

    __shared__ __attribute__((aligned(16))) float s_sim[9 * LL];
    __shared__ float s_w[81];

    const float* simb = sim + (long)b * 9 * LL;

#if __has_builtin(__builtin_amdgcn_tensor_load_to_lds)
    // --- TDM: 1-row 2D tensor, data_size=4B, 6084 elements -> LDS ---
    if (threadIdx.x == 0) {
        const unsigned int NE = 9u * LL;                       // 6084
        unsigned int lds_off = (unsigned int)(size_t)(void*)&s_sim[0]; // LDS_ADDR = flat[31:0]
        unsigned long long ga = (unsigned long long)(size_t)(const void*)simb;
        v4u g0;
        g0[0] = 1u;                                            // count=1 (valid), user mode
        g0[1] = lds_off;                                       // lds_addr
        g0[2] = (unsigned int)(ga & 0xFFFFFFFFu);              // global_addr[31:0]
        g0[3] = (unsigned int)((ga >> 32) & 0x01FFFFFFu) | (2u << 30); // ga[56:32] | type=2
        v8i_ g1;
        g1[0] = 0x00020000;                                    // data_size=2 (4B), no flags
        g1[1] = (int)((NE & 0xFFFFu) << 16);                   // tensor_dim0[15:0]
        g1[2] = (int)((NE >> 16) | (1u << 16));                // tensor_dim0[31:16] | tensor_dim1=1
        g1[3] = (int)((NE & 0xFFFFu) << 16);                   // tile_dim0 = NE
        g1[4] = 1;                                             // tile_dim1=1, tile_dim2=0
        g1[5] = (int)NE;                                       // tensor_dim0_stride[31:0]
        g1[6] = 0;                                             // stride0[47:32], dim1_stride lo
        g1[7] = 0;
        v4i_ gz = {0, 0, 0, 0};
#if __clang_major__ >= 23
        v8i_ gz8 = {0, 0, 0, 0, 0, 0, 0, 0};
        __builtin_amdgcn_tensor_load_to_lds(g0, g1, gz, gz, gz8, 0);
#else
        __builtin_amdgcn_tensor_load_to_lds(g0, g1, gz, gz, 0);
#endif
        __builtin_amdgcn_s_wait_tensorcnt(0);
    }
#else
    for (int i = threadIdx.x; i < 9 * LL; i += blockDim.x) s_sim[i] = simb[i];
#endif
    if (threadIdx.x < 81) s_w[threadIdx.x] = conv_w[o * 81 + threadIdx.x];
    __syncthreads();

    float scale = g[o] * rsqrtf(var[o] + BN_EPS);
    float shift = bta[o] - mu[o] * scale;

    for (int p = threadIdx.x; p < LL; p += blockDim.x) {
        int h = p / WW, w = p % WW;
        float acc = 0.f;
#pragma unroll
        for (int ic = 0; ic < 9; ++ic) {
#pragma unroll
            for (int ki = 0; ki < 3; ++ki) {
                int hh = h + ki - 1;
                if (hh < 0 || hh >= HH) continue;
#pragma unroll
                for (int kj = 0; kj < 3; ++kj) {
                    int ww = w + kj - 1;
                    if (ww < 0 || ww >= WW) continue;
                    acc += s_w[ic * 9 + ki * 3 + kj] * s_sim[ic * LL + hh * WW + ww];
                }
            }
        }
        float val = acc * scale + shift;
        val = (val >= 0.f) ? val : NEG_SLOPE * val;
        long idx = ((long)b * CC + o) * LL + p;
        xm[idx] = val * x[idx];
    }
}

// ---------------------------------------------------------------------------
// Generic batched bf16-WMMA GEMM:  C = alpha * opA(A) * opB(B)   (f32 accum)
//   block tile 64x128, 8 waves, 32x32 per wave via 4x v_wmma_f32_16x16x32_bf16
//   TA/TB/MODE are template params: guards fold away; edge tiles take a
//   separate uniform-branch guarded path.
// ---------------------------------------------------------------------------
#define GBM 64
#define GBN 128
#define GBK 32
#define LDT 40     // padded bf16 K-stride in LDS (80B rows -> conflict-free b128)

__device__ inline v16bf load_frag(const __bf16* p)
{
    union { v16bf v; v8bf h[2]; } u;
    u.h[0] = *(const v8bf*)p;
    u.h[1] = *(const v8bf*)(p + 16);
    return u.v;
}

template<int TA, int TB, int MODE>
__global__ __launch_bounds__(256)
void gemm_wmma_bf16(const float* __restrict__ A, const float* __restrict__ Bm,
                    float* __restrict__ C,
                    int M, int N, int K,
                    int lda, int ldb, int ldc,
                    long strideA, long strideB, long strideC,
                    float alpha,
                    const float* __restrict__ bg, const float* __restrict__ bb,
                    const float* __restrict__ bm, const float* __restrict__ bv)
{
    __shared__ __attribute__((aligned(16))) __bf16 as[GBM * LDT];
    __shared__ __attribute__((aligned(16))) __bf16 bs[GBN * LDT];

    int batch = blockIdx.z;
    A  += (long)batch * strideA;
    Bm += (long)batch * strideB;
    C  += (long)batch * strideC;

    const int tid  = threadIdx.x;
    const int wave = tid >> 5;
    const int lane = tid & 31;
    const int wrow = wave >> 2;       // 0..1
    const int wcol = wave & 3;        // 0..3
    const int hk   = lane >> 4;       // half-wave
    const int l16  = lane & 15;

    const int m0 = blockIdx.y * GBM;
    const int n0 = blockIdx.x * GBN;

    const bool fullM = (m0 + GBM <= M);
    const bool fullN = (n0 + GBN <= N);

    v8f acc[2][2] = {};

    for (int k0 = 0; k0 < K; k0 += GBK) {
        const bool fullK = (k0 + GBK <= K);

        // ---------------- stage A tile (GBM x GBK), [m][k] layout -----------
        if (TA == 0) {
            const int k4 = (tid & 7) * 4;
            if (fullM && fullK) {
#pragma unroll
                for (int pass = 0; pass < 2; ++pass) {
                    int r = (tid >> 3) + pass * 32;
                    float4 v = *(const float4*)(A + (long)(m0 + r) * lda + (k0 + k4));
                    __bf16* dst = &as[r * LDT + k4];
                    dst[0] = (__bf16)v.x; dst[1] = (__bf16)v.y;
                    dst[2] = (__bf16)v.z; dst[3] = (__bf16)v.w;
                }
            } else {
#pragma unroll
                for (int pass = 0; pass < 2; ++pass) {
                    int r = (tid >> 3) + pass * 32;
                    int gm = m0 + r;
                    float e[4] = {0.f, 0.f, 0.f, 0.f};
                    if (gm < M) {
                        const float* src = A + (long)gm * lda + (k0 + k4);
#pragma unroll
                        for (int j = 0; j < 4; ++j)
                            if (k0 + k4 + j < K) e[j] = src[j];
                    }
                    __bf16* dst = &as[r * LDT + k4];
#pragma unroll
                    for (int j = 0; j < 4; ++j) dst[j] = (__bf16)e[j];
                }
            }
        } else {  // a(m,k) = A[k*lda + m]; load 4 consecutive m
            const int m4 = (tid & 15) * 4;
            if (fullM && fullK) {
#pragma unroll
                for (int pass = 0; pass < 2; ++pass) {
                    int kk = (tid >> 4) + pass * 16;
                    float4 v = *(const float4*)(A + (long)(k0 + kk) * lda + (m0 + m4));
                    as[(m4 + 0) * LDT + kk] = (__bf16)v.x;
                    as[(m4 + 1) * LDT + kk] = (__bf16)v.y;
                    as[(m4 + 2) * LDT + kk] = (__bf16)v.z;
                    as[(m4 + 3) * LDT + kk] = (__bf16)v.w;
                }
            } else {
#pragma unroll
                for (int pass = 0; pass < 2; ++pass) {
                    int kk = (tid >> 4) + pass * 16;
                    int gk = k0 + kk;
                    float e[4] = {0.f, 0.f, 0.f, 0.f};
                    if (gk < K) {
                        const float* src = A + (long)gk * lda + (m0 + m4);
#pragma unroll
                        for (int j = 0; j < 4; ++j)
                            if (m0 + m4 + j < M) e[j] = src[j];
                    }
#pragma unroll
                    for (int j = 0; j < 4; ++j)
                        as[(m4 + j) * LDT + kk] = (__bf16)e[j];
                }
            }
        }

        // ---------------- stage B tile (GBN x GBK), [n][k] layout -----------
        if (TB == 1) {  // b(k,n) = B[n*ldb + k]; k contiguous
            const int k4 = (tid & 7) * 4;
            if (fullN && fullK) {
#pragma unroll
                for (int pass = 0; pass < 4; ++pass) {
                    int nn = (tid >> 3) + pass * 32;
                    float4 v = *(const float4*)(Bm + (long)(n0 + nn) * ldb + (k0 + k4));
                    __bf16* dst = &bs[nn * LDT + k4];
                    dst[0] = (__bf16)v.x; dst[1] = (__bf16)v.y;
                    dst[2] = (__bf16)v.z; dst[3] = (__bf16)v.w;
                }
            } else {
#pragma unroll
                for (int pass = 0; pass < 4; ++pass) {
                    int nn = (tid >> 3) + pass * 32;
                    int gn = n0 + nn;
                    float e[4] = {0.f, 0.f, 0.f, 0.f};
                    if (gn < N) {
                        const float* src = Bm + (long)gn * ldb + (k0 + k4);
#pragma unroll
                        for (int j = 0; j < 4; ++j)
                            if (k0 + k4 + j < K) e[j] = src[j];
                    }
                    __bf16* dst = &bs[nn * LDT + k4];
#pragma unroll
                    for (int j = 0; j < 4; ++j) dst[j] = (__bf16)e[j];
                }
            }
        } else {        // b(k,n) = B[k*ldb + n]; load 4 consecutive n, scatter
            const int n4 = (tid & 31) * 4;
            if (fullN && fullK) {
#pragma unroll
                for (int pass = 0; pass < 4; ++pass) {
                    int kk = (tid >> 5) + pass * 8;
                    float4 v = *(const float4*)(Bm + (long)(k0 + kk) * ldb + (n0 + n4));
                    bs[(n4 + 0) * LDT + kk] = (__bf16)v.x;
                    bs[(n4 + 1) * LDT + kk] = (__bf16)v.y;
                    bs[(n4 + 2) * LDT + kk] = (__bf16)v.z;
                    bs[(n4 + 3) * LDT + kk] = (__bf16)v.w;
                }
            } else {
#pragma unroll
                for (int pass = 0; pass < 4; ++pass) {
                    int kk = (tid >> 5) + pass * 8;
                    int gk = k0 + kk;
                    float e[4] = {0.f, 0.f, 0.f, 0.f};
                    if (gk < K) {
                        const float* src = Bm + (long)gk * ldb + (n0 + n4);
#pragma unroll
                        for (int j = 0; j < 4; ++j)
                            if (n0 + n4 + j < N) e[j] = src[j];
                    }
#pragma unroll
                    for (int j = 0; j < 4; ++j)
                        bs[(n4 + j) * LDT + kk] = (__bf16)e[j];
                }
            }
        }

        // prefetch next K-tile into GL2 (global_prefetch_b8; speculative)
        if (fullK && k0 + GBK < K) {
            if (TA == 0) {
                int gm = m0 + (tid >> 3);
                if (gm < M) __builtin_prefetch(A + (long)gm * lda + (k0 + GBK), 0, 1);
            }
            if (TB == 1) {
                int gn = n0 + (tid >> 3);
                if (gn < N) __builtin_prefetch(Bm + (long)gn * ldb + (k0 + GBK), 0, 1);
            }
        }
        __syncthreads();

        // ---------------- fragments + 4 WMMAs per wave ----------------------
        v16bf af[2], bf[2];
#pragma unroll
        for (int i = 0; i < 2; ++i)
            af[i] = load_frag(&as[(wrow * 32 + i * 16 + l16) * LDT + hk * 8]);
#pragma unroll
        for (int j = 0; j < 2; ++j)
            bf[j] = load_frag(&bs[(wcol * 32 + j * 16 + l16) * LDT + hk * 8]);

#pragma unroll
        for (int i = 0; i < 2; ++i)
#pragma unroll
            for (int j = 0; j < 2; ++j)
                acc[i][j] = __builtin_amdgcn_wmma_f32_16x16x32_bf16(
                    false, af[i], false, bf[j], (short)0, acc[i][j], false, false);
        __syncthreads();
    }

    // ---------------- epilogue / store --------------------------------------
#pragma unroll
    for (int i = 0; i < 2; ++i) {
        int gm_base = m0 + wrow * 32 + i * 16 + hk * 8;
#pragma unroll
        for (int j = 0; j < 2; ++j) {
            int gn = n0 + wcol * 32 + j * 16 + l16;
            if (!(fullM && fullN) && gn >= N) continue;
#pragma unroll
            for (int v = 0; v < 8; ++v) {
                int gm = gm_base + v;
                if (!(fullM && fullN) && gm >= M) continue;
                float val = acc[i][j][v] * alpha;
                long idx = (long)gm * ldc + gn;
                if (MODE == 1) {
                    float s = bg[gm] * rsqrtf(bv[gm] + BN_EPS);
                    float t = bb[gm] - bm[gm] * s;
                    val = val * s + t;
                    val = (val >= 0.f) ? val : NEG_SLOPE * val;
                    val += C[idx];          // + xm (X written in place)
                }
                C[idx] = val;
            }
        }
    }
}

// ---------------------------------------------------------------------------
// Kernel: GNN — one 26x26 matrix per workgroup, everything in LDS
// ---------------------------------------------------------------------------
__global__ __launch_bounds__(256)
void gnn_kernel(const float* __restrict__ X, const float* __restrict__ x,
                const float* __restrict__ Wg,
                const float* __restrict__ ln_a, const float* __restrict__ ln_b,
                float* __restrict__ out)
{
    int nc = blockIdx.x;                        // b*512 + c
    const float* Xf = X + (long)nc * LL;

    __shared__ float sX[LL], sE[LL], sXn[LL], sNode[LL], sW[LL];
    __shared__ float sCM[WW], sInv[WW], sA[WW], sB[WW];

    int t = threadIdx.x;
    for (int i = t; i < LL; i += 256) { sX[i] = Xf[i]; sW[i] = Wg[i]; }
    if (t < WW) { sA[t] = ln_a[t]; sB[t] = ln_b[t]; }
    __syncthreads();

    if (t < WW) {
        float mx = -INFINITY;
        for (int h = 0; h < HH; ++h) mx = fmaxf(mx, sX[h * WW + t]);
        float s = 0.f;
        for (int h = 0; h < HH; ++h) s += __expf(sX[h * WW + t] - mx);
        sCM[t]  = mx;
        sInv[t] = 1.f / s;
    }
    __syncthreads();
    for (int i = t; i < LL; i += 256) {
        int w = i % WW;
        sE[i] = __expf(sX[i] - sCM[w]);
    }
    if (t < HH) {
        float mean = 0.f;
        for (int w = 0; w < WW; ++w) mean += sX[t * WW + w];
        mean *= (1.f / WW);
        float var = 0.f;
        for (int w = 0; w < WW; ++w) { float d = sX[t * WW + w] - mean; var += d * d; }
        float stdv = sqrtf(var / (WW - 1));
        float inv  = 1.f / (stdv + LN_EPS);
        for (int w = 0; w < WW; ++w)
            sXn[t * WW + w] = sA[w] * (sX[t * WW + w] - mean) * inv + sB[w];
    }
    __syncthreads();
    for (int i = t; i < LL; i += 256) {
        int w = i / WW, v = i % WW;
        float acc = 0.f;
        for (int h = 0; h < HH; ++h) acc += sE[h * WW + w] * sXn[h * WW + v];
        sNode[i] = acc * sInv[w];
    }
    __syncthreads();
    for (int i = t; i < LL; i += 256) {
        int w = i / WW, u = i % WW;
        float acc = 0.f;
        for (int v = 0; v < WW; ++v) acc += sNode[w * WW + v] * sW[v * WW + u];
        acc = fmaxf(acc, 0.f);
        long idx = (long)nc * LL + i;
        out[idx] = acc + x[idx];
    }
}

// ---------------------------------------------------------------------------
// Host launch
// ---------------------------------------------------------------------------
extern "C" void kernel_launch(void* const* d_in, const int* in_sizes, int n_in,
                              void* d_out, int out_size, void* d_ws, size_t ws_size,
                              hipStream_t stream)
{
    const float* x       = (const float*)d_in[0];
    const float* conv_w  = (const float*)d_in[1];
    const float* bn1_g   = (const float*)d_in[2];
    const float* bn1_b   = (const float*)d_in[3];
    const float* bn1_m   = (const float*)d_in[4];
    const float* bn1_v   = (const float*)d_in[5];
    const float* m1_w    = (const float*)d_in[6];
    const float* m2_w    = (const float*)d_in[7];
    const float* conv2_w = (const float*)d_in[8];
    const float* bn2_g   = (const float*)d_in[9];
    const float* bn2_b   = (const float*)d_in[10];
    const float* bn2_m   = (const float*)d_in[11];
    const float* bn2_v   = (const float*)d_in[12];
    const float* gnn_W   = (const float*)d_in[13];
    const float* ln_a    = (const float*)d_in[14];
    const float* ln_b    = (const float*)d_in[15];
    float* out = (float*)d_out;

    // workspace layout (floats)
    const long SIM_N = (long)BB * 9 * LL;        //  1.6 MB
    const long XM_N  = (long)BB * CC * LL;       // 88.6 MB
    const long BM_N  = (long)BB * CQ * LL;       // 22.2 MB
    const long P_N   = (long)BB * CQ * CC;       // 16.8 MB
    float* sim = (float*)d_ws;
    float* xm  = sim + SIM_N;
    float* Bmv = xm  + XM_N;
    float* Pv  = Bmv + BM_N;
    float* Vv  = Pv  + P_N;
    float* Yv  = Vv  + P_N;                      // BB*LL*CC floats

    // 1) self-similarity
    selfsim_kernel<<<(BB * LL) / 256, 256, 0, stream>>>(x, sim);

    // 2) conv3x3 + BN1 + LeakyReLU, xm = m*x   (TDM stages sim[b] into LDS)
    conv_bn_mul_kernel<<<BB * CC, 256, 0, stream>>>(sim, conv_w,
        bn1_g, bn1_b, bn1_m, bn1_v, x, xm);

    // 3) Bm = m1_w(128x512) @ x(512x676)
    gemm_wmma_bf16<0, 0, 0><<<dim3(6, 2, BB), 256, 0, stream>>>(
        m1_w, x, Bmv, CQ, LL, CC, CC, LL, LL,
        0L, (long)CC * LL, (long)CQ * LL, 1.f,
        nullptr, nullptr, nullptr, nullptr);

    // 4) P = Bm(128x676) @ x^T(676x512)     [B transposed]
    gemm_wmma_bf16<0, 1, 0><<<dim3(4, 2, BB), 256, 0, stream>>>(
        Bmv, x, Pv, CQ, CC, LL, LL, LL, CC,
        (long)CQ * LL, (long)CC * LL, (long)CQ * CC, 1.f,
        nullptr, nullptr, nullptr, nullptr);

    // 5) V = P(128x512) @ m2_w^T(512x512) / L
    gemm_wmma_bf16<0, 1, 0><<<dim3(4, 2, BB), 256, 0, stream>>>(
        Pv, m2_w, Vv, CQ, CC, CC, CC, CC, CC,
        (long)CQ * CC, 0L, (long)CQ * CC, 1.f / (float)LL,
        nullptr, nullptr, nullptr, nullptr);

    // 6) Y = Bm^T(676x128) @ V(128x512)     [A transposed]
    gemm_wmma_bf16<1, 0, 0><<<dim3(4, 11, BB), 256, 0, stream>>>(
        Bmv, Vv, Yv, LL, CC, CQ, LL, CC, CC,
        (long)CQ * LL, (long)CQ * CC, (long)LL * CC, 1.f,
        nullptr, nullptr, nullptr, nullptr);

    // 7) X = lrelu(bn2(conv2_w @ view(Y,(512,676)))) + xm   (written into xm)
    gemm_wmma_bf16<0, 0, 1><<<dim3(6, 8, BB), 256, 0, stream>>>(
        conv2_w, Yv, xm, CC, LL, CC, CC, LL, LL,
        0L, (long)CC * LL, (long)CC * LL, 1.f,
        bn2_g, bn2_b, bn2_m, bn2_v);

    // 8) GNN + residual
    gnn_kernel<<<BB * CC, 256, 0, stream>>>(xm, x, gnn_W, ln_a, ln_b, out);
}
